// PointNet2ClsSsg_86363202388414
// MI455X (gfx1250) — matrix operand order repaired and checked by reading, not compiled, not run on hardware
//
#include <hip/hip_runtime.h>
#include <hip/hip_bf16.h>

typedef __attribute__((ext_vector_type(16))) _Float16 v16h;
typedef __attribute__((ext_vector_type(8)))  _Float16 v8h;
typedef __attribute__((ext_vector_type(8)))  float    v8f;

#define EPS_BN 1e-5f

// ---------------------------------------------------------------------------
// Farthest point sampling: one block per batch, dist[] resident in LDS.
// Matches reference scan semantics: fid[0]=0, then argmax of running min-dist.
// ---------------------------------------------------------------------------
__global__ __launch_bounds__(256) void fps_kernel(const float* __restrict__ xyz,
                                                  int N, int npoint,
                                                  int* __restrict__ fid) {
    __shared__ float dist[4096];
    __shared__ float rmax[256];
    __shared__ int   ridx[256];
    __shared__ int   s_far;
    const int b = blockIdx.x;
    const int t = threadIdx.x;
    const float* xb = xyz + (size_t)b * N * 3;

    for (int i = t; i < N; i += 256) dist[i] = 1e10f;
    if (t == 0) s_far = 0;
    __syncthreads();

    for (int it = 0; it < npoint; ++it) {
        const int far = s_far;
        if (t == 0) fid[b * npoint + it] = far;
        const float cx = xb[far * 3 + 0];
        const float cy = xb[far * 3 + 1];
        const float cz = xb[far * 3 + 2];
        float bm = -1.0f; int bi = 0;
        for (int i = t; i < N; i += 256) {
            const float dx = xb[i * 3 + 0] - cx;
            const float dy = xb[i * 3 + 1] - cy;
            const float dz = xb[i * 3 + 2] - cz;
            const float d  = dx * dx + dy * dy + dz * dz;
            const float nd = fminf(dist[i], d);
            dist[i] = nd;
            if (nd > bm) { bm = nd; bi = i; }
        }
        rmax[t] = bm; ridx[t] = bi;
        __syncthreads();
        for (int s = 128; s > 0; s >>= 1) {
            if (t < s) {
                const float om = rmax[t + s]; const int oi = ridx[t + s];
                if (om > rmax[t] || (om == rmax[t] && oi < ridx[t])) {
                    rmax[t] = om; ridx[t] = oi;
                }
            }
            __syncthreads();
        }
        if (t == 0) s_far = ridx[0];
        __syncthreads();
    }
}

// ---------------------------------------------------------------------------
// Ball query: one wave32 per query point. Ballot + prefix-popcount compaction
// produces the first `ns` in-radius indices in ascending order (== reference
// where/sort/slice). Remaining slots filled with the first index.
// ---------------------------------------------------------------------------
__global__ __launch_bounds__(256) void ball_query_kernel(
    const float* __restrict__ xyz, const int* __restrict__ fid,
    float* __restrict__ new_xyz, int* __restrict__ idx,
    int N, int S, int ns, float r2) {
    const int wq   = blockIdx.x * 8 + (threadIdx.x >> 5);
    const int lane = threadIdx.x & 31;
    const int b = wq / S, s = wq % S;
    const float* xb = xyz + (size_t)b * N * 3;
    const int f = fid[b * S + s];
    const float cx = xb[f * 3 + 0], cy = xb[f * 3 + 1], cz = xb[f * 3 + 2];
    if (lane < 3) new_xyz[((size_t)b * S + s) * 3 + lane] = xb[f * 3 + lane];
    int* ob = idx + ((size_t)b * S + s) * ns;

    int cnt = 0, first = -1;
    for (int base = 0; base < N && cnt < ns; base += 32) {
        const int i = base + lane;
        bool in = false;
        if (i < N) {
            const float dx = xb[i * 3 + 0] - cx;
            const float dy = xb[i * 3 + 1] - cy;
            const float dz = xb[i * 3 + 2] - cz;
            in = (dx * dx + dy * dy + dz * dz) <= r2;
        }
        const unsigned m = (unsigned)__ballot(in);
        if (first < 0 && m) first = base + __ffs(m) - 1;
        const int pos = cnt + __popc(m & ((1u << lane) - 1u));
        if (in && pos < ns) ob[pos] = i;
        cnt += __popc(m);
    }
    if (first < 0) first = 0;
    const int valid = cnt < ns ? cnt : ns;
    for (int p = lane; p < ns; p += 32)
        if (p >= valid) ob[p] = first;
}

// ---------------------------------------------------------------------------
// Build grouped input rows (f16): [xyz(-center) | point features | zero pad].
// idx==nullptr -> identity gather (group_all). One thread per element.
// ---------------------------------------------------------------------------
__global__ __launch_bounds__(256) void group_kernel(
    const float* __restrict__ xyz, const _Float16* __restrict__ pts,
    const int* __restrict__ idx, const float* __restrict__ new_xyz,
    _Float16* __restrict__ G, int N, int S, int ns, int C, int Kpad,
    int center, long total) {
    const long tid = (long)blockIdx.x * blockDim.x + threadIdx.x;
    if (tid >= total) return;
    const int row = (int)(tid / Kpad);
    const int k   = (int)(tid % Kpad);
    const int b   = row / (S * ns);
    const int rem = row % (S * ns);
    const int s   = rem / ns;
    const int j   = rem % ns;
    const int id  = idx ? idx[((size_t)b * S + s) * ns + j] : j;
    _Float16 v = (_Float16)0.0f;
    if (k < 3) {
        float f = xyz[((size_t)b * N + id) * 3 + k];
        if (center) f -= new_xyz[((size_t)b * S + s) * 3 + k];
        v = (_Float16)f;
    } else if (k < 3 + C) {
        v = pts[((size_t)b * N + id) * C + (k - 3)];
    }
    G[tid] = v;
}

// ---------------------------------------------------------------------------
// Pre-pad + convert a weight matrix (Nw x Kw, f32) into f16 (Npad x Kpad),
// zero-filling the padding so the GEMM inner loop needs no bounds checks.
// ---------------------------------------------------------------------------
__global__ __launch_bounds__(256) void convert_weights_kernel(
    const float* __restrict__ W, _Float16* __restrict__ W16,
    int Kw, int Kpad, int Nw, long total) {
    const long tid = (long)blockIdx.x * blockDim.x + threadIdx.x;
    if (tid >= total) return;
    const int n = (int)(tid / Kpad);
    const int k = (int)(tid % Kpad);
    W16[tid] = (_Float16)((n < Nw && k < Kw) ? W[(size_t)n * Kw + k] : 0.0f);
}

// ---------------------------------------------------------------------------
// WMMA GEMM: Out[M x Nw] = act(A) [M x K] * W16^T + bias,
// f16 multiply / f32 accumulate via v_wmma_f32_16x16x32_f16.
//
// Block tile 128(M) x 64(N): 8 waves, each wave owns 16 rows x 64 cols
// = 4 WMMA accumulators, so one A fragment (incl. fused BN+ReLU transform)
// feeds 4 WMMAs. W16 is pre-padded f16, so the 32x64 weight tile staging is
// pure aligned 16B global_load -> ds_store traffic, and B fragments are
// two 16B LDS runs 32B apart (ds_load_b128 pairs). BN scale/shift vectors are
// staged to LDS once at entry so the inner loop issues no scale/shift VMEM.
// TRANSFORM=true : A is f32, per-input-channel BN scale/shift + ReLU fused.
// TRANSFORM=false: A is f16, loaded directly in fragment order.
// Per-output-channel BN statistics reduced in LDS, then global f32 atomics.
// ---------------------------------------------------------------------------
template <bool TRANSFORM>
__global__ __launch_bounds__(256) void wmma_gemm_kernel(
    const void* __restrict__ Av, int K,
    const _Float16* __restrict__ W16,
    const float* __restrict__ bias,
    const float* __restrict__ in_scale, const float* __restrict__ in_shift,
    float* __restrict__ Out, int M, int Nw,
    float* __restrict__ bn_sum, float* __restrict__ bn_sumsq) {
    __shared__ _Float16 sB[4][16][32];   // [n-subtile][col within 16][k] : 4KB
    __shared__ float s_scale[1024];      // BN transform of the input channels
    __shared__ float s_shift[1024];
    __shared__ float s_sum[64];
    __shared__ float s_sq[64];
    if (threadIdx.x < 64) { s_sum[threadIdx.x] = 0.0f; s_sq[threadIdx.x] = 0.0f; }
    if (TRANSFORM) {
        for (int i = threadIdx.x; i < K; i += 256) {
            s_scale[i] = in_scale[i];
            s_shift[i] = in_shift[i];
        }
    }

    const int wave = threadIdx.x >> 5;
    const int lane = threadIdx.x & 31;
    const int half = lane >> 4;     // 0: lanes 0-15, 1: lanes 16-31
    const int r    = lane & 15;
    const int m0   = blockIdx.x * 128 + wave * 16;
    const int nblk = blockIdx.y * 64;

    const int  rowA  = m0 + r;      // A-fragment row for this lane (M = lane&15)
    const bool rowOK = rowA < M;

    // staging coordinates: thread moves 8 consecutive k (16B) of one (nt, col)
    const int e   = threadIdx.x * 8;      // 0..2047
    const int snt = e >> 9;               // n-subtile 0..3
    const int sc  = (e >> 5) & 15;        // column within subtile
    const int sk  = e & 31;               // k start (multiple of 8)
    const _Float16* wsrc = W16 + (size_t)(nblk + snt * 16 + sc) * K + sk;

    v8f acc[4] = {};
    __syncthreads();

    for (int k0 = 0; k0 < K; k0 += 32) {
        // ---- stage 32x64 pre-padded f16 weight tile into LDS ----
        *(v8h*)&sB[snt][sc][sk] = *(const v8h*)(wsrc + k0);
        __syncthreads();

        // ---- A fragment (16x32 f16): lane<16 -> K=[k0..k0+7],[k0+16..k0+23]
        //                              lane>=16 -> +8 offset on both spans ----
        v16h a;
        if (rowOK) {
            if (TRANSFORM) {
                const float* ap = (const float*)Av + (size_t)rowA * K + k0 + half * 8;
                __builtin_prefetch(ap + 32, 0, 3);   // next K tile
                const int kbase = k0 + half * 8;
#pragma unroll
                for (int j = 0; j < 8; ++j) {
                    a[j]     = (_Float16)fmaxf(ap[j]      * s_scale[kbase + j]
                                               + s_shift[kbase + j], 0.0f);
                    a[8 + j] = (_Float16)fmaxf(ap[16 + j] * s_scale[kbase + 16 + j]
                                               + s_shift[kbase + 16 + j], 0.0f);
                }
            } else {
                const _Float16* ap = (const _Float16*)Av + (size_t)rowA * K + k0 + half * 8;
                const v8h lo = *(const v8h*)ap;
                const v8h hi = *(const v8h*)(ap + 16);
#pragma unroll
                for (int j = 0; j < 8; ++j) { a[j] = lo[j]; a[8 + j] = hi[j]; }
            }
        } else {
#pragma unroll
            for (int j = 0; j < 16; ++j) a[j] = (_Float16)0.0f;
        }

        // ---- 4 WMMAs sharing one A fragment; B fragments from LDS ----
        // B layout: lanes 0-15 hold K=k0..k0+7,k0+16..k0+23 of col n, lanes 16-31 +8.
#pragma unroll
        for (int nt = 0; nt < 4; ++nt) {
            const _Float16* bp = &sB[nt][r][half * 8];  // two 8-half runs
            const v8h blo = *(const v8h*)bp;
            const v8h bhi = *(const v8h*)(bp + 16);
            v16h bfrag;
#pragma unroll
            for (int j = 0; j < 8; ++j) { bfrag[j] = blo[j]; bfrag[8 + j] = bhi[j]; }
            acc[nt] = __builtin_amdgcn_wmma_f32_16x16x32_f16(
                /*neg_a=*/false, a, /*neg_b=*/false, bfrag,
                /*c_mod=*/(short)0, acc[nt], /*reuse_a=*/false, /*reuse_b=*/false);
        }
        __syncthreads();   // before next stage overwrites sB
    }

    // ---- epilogue: bias, store, per-channel BN statistics ----
#pragma unroll
    for (int nt = 0; nt < 4; ++nt) {
        const int  n     = nblk + nt * 16 + r;
        const bool colOK = n < Nw;
        const float bv   = colOK ? bias[n] : 0.0f;
        float lsum = 0.0f, lsq = 0.0f;
#pragma unroll
        for (int rr = 0; rr < 8; ++rr) {
            const int row = m0 + rr + half * 8;   // D: VGPR rr -> M = rr + 8*half
            const float v = acc[nt][rr] + bv;
            if (row < M && colOK) {
                Out[(size_t)row * Nw + n] = v;
                lsum += v; lsq += v * v;
            }
        }
        if (bn_sum && colOK) {
            atomicAdd(&s_sum[nt * 16 + r], lsum);
            atomicAdd(&s_sq[nt * 16 + r], lsq);
        }
    }
    if (bn_sum) {
        __syncthreads();
        if (threadIdx.x < 64 && (nblk + threadIdx.x) < Nw) {
            atomicAdd(&bn_sum[nblk + threadIdx.x],   s_sum[threadIdx.x]);
            atomicAdd(&bn_sumsq[nblk + threadIdx.x], s_sq[threadIdx.x]);
        }
    }
}

// ---------------------------------------------------------------------------
// Fold batch statistics into per-channel scale/shift (population variance).
// ---------------------------------------------------------------------------
__global__ __launch_bounds__(256) void bn_finalize_kernel(
    const float* __restrict__ sum, const float* __restrict__ sq, float invM,
    const float* __restrict__ gamma, const float* __restrict__ beta,
    float* __restrict__ scale, float* __restrict__ shift, int N) {
    const int i = blockIdx.x * blockDim.x + threadIdx.x;
    if (i >= N) return;
    const float mean = sum[i] * invM;
    const float var  = sq[i] * invM - mean * mean;
    const float s    = gamma[i] * rsqrtf(var + EPS_BN);
    scale[i] = s;
    shift[i] = beta[i] - mean * s;
}

// ---------------------------------------------------------------------------
// BN + ReLU + max over the neighborhood (last layer of each SA block).
// Output in f16 (it is only ever consumed as WMMA A-operand).
// ---------------------------------------------------------------------------
__global__ __launch_bounds__(256) void bn_relu_maxpool_kernel(
    const float* __restrict__ X, const float* __restrict__ scale,
    const float* __restrict__ shift, _Float16* __restrict__ out,
    int ns, int N) {
    const int c = blockIdx.y * blockDim.x + threadIdx.x;
    const int g = blockIdx.x;
    if (c >= N) return;
    const float* xp = X + (size_t)g * ns * N + c;
    const float sc = scale[c], sh = shift[c];
    float m = 0.0f;                 // ReLU output is >= 0
    for (int j = 0; j < ns; ++j)
        m = fmaxf(m, fmaxf(xp[(size_t)j * N] * sc + sh, 0.0f));
    out[(size_t)g * N + c] = (_Float16)m;
}

// ---------------------------------------------------------------------------
// Host orchestration
// ---------------------------------------------------------------------------
extern "C" void kernel_launch(void* const* d_in, const int* in_sizes, int n_in,
                              void* d_out, int out_size, void* d_ws, size_t ws_size,
                              hipStream_t stream) {
    (void)in_sizes; (void)n_in; (void)out_size; (void)ws_size;
    const int B = 16;
    const float* pc = (const float*)d_in[0];
    auto P = [&](int i) { return (const float*)d_in[i]; };
    // Input order: pc, sa1(3 x [w,b,gm,bt]), sa2, sa3, fc1(w,b), fc2, fc3, bn1(g,b), bn2(g,b)

    // ---- workspace carving (256B aligned) ----
    char* wp = (char*)d_ws;
    auto carve = [&](size_t bytes) -> void* {
        void* r = (void*)wp;
        wp += (bytes + 255) & ~(size_t)255;
        return r;
    };
    int*       fid1   = (int*)      carve((size_t)B * 512 * 4);
    float*     l1_xyz = (float*)    carve((size_t)B * 512 * 3 * 4);
    int*       idx1   = (int*)      carve((size_t)B * 512 * 32 * 4);
    int*       fid2   = (int*)      carve((size_t)B * 128 * 4);
    float*     l2_xyz = (float*)    carve((size_t)B * 128 * 3 * 4);
    int*       idx2   = (int*)      carve((size_t)B * 128 * 64 * 4);
    _Float16*  G      = (_Float16*) carve((size_t)131072 * 160 * 2);  // max grouped
    float*     bufA   = (float*)    carve((size_t)262144 * 128 * 4);  // ping
    float*     bufB   = (float*)    carve((size_t)262144 * 64 * 4);   // pong
    _Float16*  l1_pts = (_Float16*) carve((size_t)B * 512 * 128 * 2);
    _Float16*  l2_pts = (_Float16*) carve((size_t)B * 128 * 256 * 2);
    _Float16*  l3_pts = (_Float16*) carve((size_t)B * 1024 * 2);
    _Float16*  W16    = (_Float16*) carve((size_t)1024 * 1024 * 2);   // padded weights
    float*     sums   = (float*)    carve((size_t)2 * 1024 * 4);      // sum|sumsq
    float*     sumsq  = sums + 1024;
    float*     scale  = (float*)    carve((size_t)1024 * 4);
    float*     shift  = (float*)    carve((size_t)1024 * 4);

    auto gemm = [&](const void* Ain, int K, const float* Wm, int Kw,
                    const float* bias, float* out, int M, int Nw,
                    bool withSums, bool transform) {
        const int Npad = ((Nw + 63) / 64) * 64;
        const long wtotal = (long)Npad * K;
        convert_weights_kernel<<<(unsigned)((wtotal + 255) / 256), 256, 0, stream>>>(
            Wm, W16, Kw, K, Nw, wtotal);
        dim3 grid((M + 127) / 128, Npad / 64);
        float *s = nullptr, *q = nullptr;
        if (withSums) {
            hipMemsetAsync(sums, 0, 2 * 1024 * sizeof(float), stream);
            s = sums; q = sumsq;
        }
        if (transform)
            wmma_gemm_kernel<true><<<grid, 256, 0, stream>>>(
                Ain, K, W16, bias, scale, shift, out, M, Nw, s, q);
        else
            wmma_gemm_kernel<false><<<grid, 256, 0, stream>>>(
                Ain, K, W16, bias, nullptr, nullptr, out, M, Nw, s, q);
    };
    auto finalize = [&](float invM, const float* gm, const float* bt, int N) {
        bn_finalize_kernel<<<(N + 255) / 256, 256, 0, stream>>>(
            sums, sumsq, invM, gm, bt, scale, shift, N);
    };
    auto group = [&](const float* xyz, const _Float16* pts, const int* idx,
                     const float* nxyz, int N, int S, int ns, int C, int Kpad,
                     int center) {
        long total = (long)B * S * ns * Kpad;
        group_kernel<<<(unsigned)((total + 255) / 256), 256, 0, stream>>>(
            xyz, pts, idx, nxyz, G, N, S, ns, C, Kpad, center, total);
    };

    // ================= SA1: N=4096 -> S=512, r=0.2, ns=32, 3->64->64->128 ====
    fps_kernel<<<B, 256, 0, stream>>>(pc, 4096, 512, fid1);
    ball_query_kernel<<<(B * 512) / 8, 256, 0, stream>>>(
        pc, fid1, l1_xyz, idx1, 4096, 512, 32, 0.2f * 0.2f);
    group(pc, nullptr, idx1, l1_xyz, 4096, 512, 32, 0, 32, 1);
    {
        const int M = B * 512 * 32; const float invM = 1.0f / (float)M;
        gemm(G, 32, P(1), 3, P(2), bufA, M, 64, true, false);
        finalize(invM, P(3), P(4), 64);
        gemm(bufA, 64, P(5), 64, P(6), bufB, M, 64, true, true);
        finalize(invM, P(7), P(8), 64);
        gemm(bufB, 64, P(9), 64, P(10), bufA, M, 128, true, true);
        finalize(invM, P(11), P(12), 128);
        bn_relu_maxpool_kernel<<<dim3(B * 512, 1), 256, 0, stream>>>(
            bufA, scale, shift, l1_pts, 32, 128);
    }

    // ================= SA2: N=512 -> S=128, r=0.4, ns=64, 131->128->128->256 =
    fps_kernel<<<B, 256, 0, stream>>>(l1_xyz, 512, 128, fid2);
    ball_query_kernel<<<(B * 128) / 8, 256, 0, stream>>>(
        l1_xyz, fid2, l2_xyz, idx2, 512, 128, 64, 0.4f * 0.4f);
    group(l1_xyz, l1_pts, idx2, l2_xyz, 512, 128, 64, 128, 160, 1);
    {
        const int M = B * 128 * 64; const float invM = 1.0f / (float)M;
        gemm(G, 160, P(13), 131, P(14), bufA, M, 128, true, false);
        finalize(invM, P(15), P(16), 128);
        gemm(bufA, 128, P(17), 128, P(18), bufB, M, 128, true, true);
        finalize(invM, P(19), P(20), 128);
        gemm(bufB, 128, P(21), 128, P(22), bufA, M, 256, true, true);
        finalize(invM, P(23), P(24), 256);
        bn_relu_maxpool_kernel<<<dim3(B * 128, 1), 256, 0, stream>>>(
            bufA, scale, shift, l2_pts, 64, 256);
    }

    // ================= SA3 (group_all): 259->256->512->1024, maxpool 128 =====
    group(l2_xyz, l2_pts, nullptr, nullptr, 128, 1, 128, 256, 288, 0);
    {
        const int M = B * 128; const float invM = 1.0f / (float)M;
        gemm(G, 288, P(25), 259, P(26), bufA, M, 256, true, false);
        finalize(invM, P(27), P(28), 256);
        gemm(bufA, 256, P(29), 256, P(30), bufB, M, 512, true, true);
        finalize(invM, P(31), P(32), 512);
        gemm(bufB, 512, P(33), 512, P(34), bufA, M, 1024, true, true);
        finalize(invM, P(35), P(36), 1024);
        bn_relu_maxpool_kernel<<<dim3(B, 4), 256, 0, stream>>>(
            bufA, scale, shift, l3_pts, 128, 1024);
    }

    // ================= FC head: 1024->512 (bn1) ->256 (bn2) ->40 =============
    {
        const float invB = 1.0f / (float)B;
        gemm(l3_pts, 1024, P(37), 1024, P(38), bufA, B, 512, true, false);
        finalize(invB, P(43), P(44), 512);
        gemm(bufA, 512, P(39), 512, P(40), bufB, B, 256, true, true);
        finalize(invB, P(45), P(46), 256);
        gemm(bufB, 256, P(41), 256, P(42), (float*)d_out, B, 40, false, true);
    }
}
